// BatchAdditiveAttention_5171140625170
// MI455X (gfx1250) — matrix operand order repaired and checked by reading, not compiled, not run on hardware
//
#include <hip/hip_runtime.h>

// BatchAdditiveAttention for MI455X (gfx1250, wave32).
// feature:(8,20000,256) f32, emb:(8,20000,4,256) f32, w1/w2:(128,256) f32, m:(1,128) f32
// out:(8,20000,1,256) f32
//
// The 5 per-token (256->128) projections form one (800k x 256)x(256x128) GEMM
// (~52 GFLOP) -> v_wmma_f32_16x16x32_f16. Weights are converted to f16 and
// pre-swizzled into LDS B-fragments ONCE per block (128 KB): the hot loop feeds
// B with two conflict-free ds_load_b128 per WMMA, explicitly double-buffered
// (depth 2) so each WMMA waits on an OLD load pair while younger pairs stream.
// tanh/softmax use v_exp_f32 + v_rcp_f32; beta is broadcast with wave shuffles.

typedef __attribute__((ext_vector_type(16))) _Float16 v16h;
typedef __attribute__((ext_vector_type(8)))  _Float16 v8h;
typedef __attribute__((ext_vector_type(8)))  float    v8f;

#define DD   256   // input dim (d == d1)
#define DOUT 128   // d2
#define NR   4     // type axis

// A fragment (16x32 f16, wave32): lane l -> row M=l&15, half h=l>>4 holds
// K = h*16+{0..15}, packed sequentially.
__device__ __forceinline__ v16h pack_a(const float* __restrict__ row, int k0, int hf) {
  const float* p = row + k0 + hf * 16;
  float4 x0 = *(const float4*)(p + 0);
  float4 x1 = *(const float4*)(p + 4);
  float4 y0 = *(const float4*)(p + 8);
  float4 y1 = *(const float4*)(p + 12);
  v16h a;
  a[0]  = (_Float16)x0.x; a[1]  = (_Float16)x0.y; a[2]  = (_Float16)x0.z; a[3]  = (_Float16)x0.w;
  a[4]  = (_Float16)x1.x; a[5]  = (_Float16)x1.y; a[6]  = (_Float16)x1.z; a[7]  = (_Float16)x1.w;
  a[8]  = (_Float16)y0.x; a[9]  = (_Float16)y0.y; a[10] = (_Float16)y0.z; a[11] = (_Float16)y0.w;
  a[12] = (_Float16)y1.x; a[13] = (_Float16)y1.y; a[14] = (_Float16)y1.z; a[15] = (_Float16)y1.w;
  return a;
}

// Fetch a pre-swizzled B fragment from LDS: two conflict-free ds_load_b128.
__device__ __forceinline__ v16h fetch_b(const _Float16* fb, int lane) {
  v8h lo = *(const v8h*)(fb + lane * 8);
  v8h hi = *(const v8h*)(fb + 256 + lane * 8);
  return __builtin_shufflevector(lo, hi, 0, 1, 2, 3, 4, 5, 6, 7,
                                 8, 9, 10, 11, 12, 13, 14, 15);
}

// tanh(x) = 1 - 2/(exp(2x)+1); exact saturation at +-1.
__device__ __forceinline__ float fast_tanh(float x) {
  float e = __expf(2.0f * x);
  return 1.0f - 2.0f * __builtin_amdgcn_rcpf(e + 1.0f);
}

__global__ __launch_bounds__(256) void baa_kernel(
    const float* __restrict__ feature, const float* __restrict__ emb,
    const float* __restrict__ w1, const float* __restrict__ w2,
    const float* __restrict__ mvec, float* __restrict__ out, int nTok)
{
  // Pre-swizzled f16 B-fragments: frag (mat, nt, kk) = 512 halves (1 KB).
  // Blob layout: [lane*8 + e] = halves e0..7, [256 + lane*8 + e] = halves e8..15
  // for lane's (col = nt*16 + (l&15), kb = kk*32 + (l>>4)*16).
  __shared__ __align__(16) _Float16 wfrag[128][512];   // 128 KB

  const int lane = threadIdx.x & 31;
  const int wave = threadIdx.x >> 5;
  const int hf   = lane >> 4;
  const int l15  = lane & 15;

  // ---------- stage w1/w2 as f16 B-fragments in LDS (whole block) ----------
  #pragma unroll
  for (int i = 0; i < 16; ++i) {
    int frag = wave * 16 + i;                 // 0..127
    int f    = frag & 63;                     // nt*8 + kk
    int nt = f >> 3, kk = f & 7;
    const float* w = (frag < 64) ? w1 : w2;
    const float* p = w + (nt * 16 + l15) * DD + kk * 32 + hf * 16;
    float4 x0 = *(const float4*)(p + 0);
    float4 x1 = *(const float4*)(p + 4);
    float4 x2 = *(const float4*)(p + 8);
    float4 x3 = *(const float4*)(p + 12);
    v8h lo, hi;
    lo[0] = (_Float16)x0.x; lo[1] = (_Float16)x0.y; lo[2] = (_Float16)x0.z; lo[3] = (_Float16)x0.w;
    lo[4] = (_Float16)x1.x; lo[5] = (_Float16)x1.y; lo[6] = (_Float16)x1.z; lo[7] = (_Float16)x1.w;
    hi[0] = (_Float16)x2.x; hi[1] = (_Float16)x2.y; hi[2] = (_Float16)x2.z; hi[3] = (_Float16)x2.w;
    hi[4] = (_Float16)x3.x; hi[5] = (_Float16)x3.y; hi[6] = (_Float16)x3.z; hi[7] = (_Float16)x3.w;
    *(v8h*)(&wfrag[frag][lane * 8])       = lo;
    *(v8h*)(&wfrag[frag][256 + lane * 8]) = hi;
  }
  __syncthreads();

  const long tokBase = ((long)blockIdx.x * 8 + wave) * 16;
  if (tokBase >= nTok) return;   // no barriers below this point

  // m[nt*16 + l15] for the 8 N-tiles (this lane's D-fragment column)
  float mv[8];
  #pragma unroll
  for (int nt = 0; nt < 8; ++nt) mv[nt] = mvec[nt * 16 + l15];

  // ---------- f_proj = feature_tile (16x256) @ w1^T -> 8 C-frags ----------
  const float* frow = feature + (size_t)(tokBase + l15) * DD;
  v8f accF[8];
  #pragma unroll
  for (int nt = 0; nt < 8; ++nt) { v8f z = {}; accF[nt] = z; }
  #pragma unroll
  for (int kk = 0; kk < 8; ++kk) {
    v16h a  = pack_a(frow, kk * 32, hf);
    v16h b0 = fetch_b(&wfrag[0 * 8 + kk][0], lane);   // depth-2 prefetch pipeline
    v16h b1 = fetch_b(&wfrag[1 * 8 + kk][0], lane);
    #pragma unroll
    for (int nt = 0; nt < 8; ++nt) {
      v16h bn = (nt < 6) ? fetch_b(&wfrag[(nt + 2) * 8 + kk][0], lane) : b1;
      accF[nt] = __builtin_amdgcn_wmma_f32_16x16x32_f16(
          false, a, false, b0, (short)0, accF[nt], false, false);
      b0 = b1; b1 = bn;
    }
  }

  // ---------- per type r: t_proj (same pipelined shape), tanh, dot m ----------
  float sc[NR][8];  // sc[r][v] = score of token (hf*8+v) for type r
  #pragma unroll
  for (int r = 0; r < NR; ++r) {
    const float* erow = emb + ((size_t)(tokBase + l15) * NR + r) * DD;

    v8f acc[8];
    #pragma unroll
    for (int nt = 0; nt < 8; ++nt) { v8f z = {}; acc[nt] = z; }
    #pragma unroll
    for (int kk = 0; kk < 8; ++kk) {
      v16h a  = pack_a(erow, kk * 32, hf);            // one A frag live at a time
      v16h b0 = fetch_b(&wfrag[64 + 0 * 8 + kk][0], lane);
      v16h b1 = fetch_b(&wfrag[64 + 1 * 8 + kk][0], lane);
      #pragma unroll
      for (int nt = 0; nt < 8; ++nt) {
        v16h bn = (nt < 6) ? fetch_b(&wfrag[64 + (nt + 2) * 8 + kk][0], lane) : b1;
        acc[nt] = __builtin_amdgcn_wmma_f32_16x16x32_f16(
            false, a, false, b0, (short)0, acc[nt], false, false);
        b0 = b1; b1 = bn;
      }
    }

    v8f partial = {};
    #pragma unroll
    for (int nt = 0; nt < 8; ++nt) {
      #pragma unroll
      for (int v = 0; v < 8; ++v) {
        float q = fast_tanh(accF[nt][v] + acc[nt][v]);  // q[token][N=nt*16+l15]
        partial[v] += q * mv[nt];
      }
    }
    // reduce over the 16 lanes of each half (xor of bits 0..3 stays in-half)
    #pragma unroll
    for (int v = 0; v < 8; ++v) {
      float x = partial[v];
      x += __shfl_xor(x, 1, 32);
      x += __shfl_xor(x, 2, 32);
      x += __shfl_xor(x, 4, 32);
      x += __shfl_xor(x, 8, 32);
      sc[r][v] = x;  // all 16 lanes of the half now hold the full score
    }
  }

  // ---------- softmax over r (all lanes, redundantly per half) ----------
  float beta0[8], beta1[8], beta2[8], beta3[8];
  #pragma unroll
  for (int v = 0; v < 8; ++v) {
    float s0 = sc[0][v], s1 = sc[1][v], s2 = sc[2][v], s3 = sc[3][v];
    float mx = fmaxf(fmaxf(s0, s1), fmaxf(s2, s3));
    float e0 = __expf(s0 - mx), e1 = __expf(s1 - mx);
    float e2 = __expf(s2 - mx), e3 = __expf(s3 - mx);
    float inv = __builtin_amdgcn_rcpf(e0 + e1 + e2 + e3);
    beta0[v] = e0 * inv; beta1[v] = e1 * inv;
    beta2[v] = e2 * inv; beta3[v] = e3 * inv;
  }

  // ---------- fused = sum_r beta[r] * emb[:, r, :]  (emb re-read hits L2) ----------
  #pragma unroll
  for (int t = 0; t < 16; ++t) {
    const long tok = tokBase + t;
    const int src = (t >> 3) * 16;     // lane 0 holds tokens 0..7, lane 16 -> 8..15
    const int v   = t & 7;
    float b0 = __shfl(beta0[v], src, 32);
    float b1 = __shfl(beta1[v], src, 32);
    float b2 = __shfl(beta2[v], src, 32);
    float b3 = __shfl(beta3[v], src, 32);
    float bb[4] = {b0, b1, b2, b3};
    const float* ep = emb + (size_t)tok * NR * DD + lane * 8;  // 8 floats per lane
    float4 o0 = make_float4(0.f, 0.f, 0.f, 0.f);
    float4 o1 = make_float4(0.f, 0.f, 0.f, 0.f);
    #pragma unroll
    for (int r = 0; r < 4; ++r) {
      float4 a0 = *(const float4*)(ep + r * DD);
      float4 a1 = *(const float4*)(ep + r * DD + 4);
      o0.x = fmaf(bb[r], a0.x, o0.x); o0.y = fmaf(bb[r], a0.y, o0.y);
      o0.z = fmaf(bb[r], a0.z, o0.z); o0.w = fmaf(bb[r], a0.w, o0.w);
      o1.x = fmaf(bb[r], a1.x, o1.x); o1.y = fmaf(bb[r], a1.y, o1.y);
      o1.z = fmaf(bb[r], a1.z, o1.z); o1.w = fmaf(bb[r], a1.w, o1.w);
    }
    float* op = out + (size_t)tok * DD + lane * 8;
    *(float4*)(op + 0) = o0;
    *(float4*)(op + 4) = o1;
  }
}

extern "C" void kernel_launch(void* const* d_in, const int* in_sizes, int n_in,
                              void* d_out, int out_size, void* d_ws, size_t ws_size,
                              hipStream_t stream) {
  const float* feature = (const float*)d_in[0];
  const float* emb     = (const float*)d_in[1];
  const float* w1      = (const float*)d_in[2];
  const float* w2      = (const float*)d_in[3];
  const float* mvec    = (const float*)d_in[4];
  float* out = (float*)d_out;

  const int nTok = in_sizes[0] / DD;          // 8*20000 = 160000
  const int tokensPerBlock = 16 * 8;          // 16 tokens/wave * 8 waves
  const int blocks = (nTok + tokensPerBlock - 1) / tokensPerBlock;

  baa_kernel<<<blocks, 256, 0, stream>>>(feature, emb, w1, w2, mvec, out, nTok);
}
